// LocalAttention_13082470384196
// MI455X (gfx1250) — matrix operand order repaired and checked by reading, not compiled, not run on hardware
//
#include <hip/hip_runtime.h>

#define BB 16
#define SS 100
#define DD 8192
#define PP 100
#define TT 102                      // padded sequence length
#define ROWS (BB * TT)              // 1632 = 102 M-tiles of 16
#define KSTEPS (DD / 32)            // 256
#define NT 7                        // N-tiles per weight (ceil(100/16))
#define NWT (3 * NT)                // 21 weight-tiles
#define BFRAG_DW 512                // dwords per (wt, kstep) B block (2KB)
#define BWS_DW ((size_t)NWT * KSTEPS * BFRAG_DW)   // 10.5 MB of bf16 hi/lo B fragments

typedef __bf16 bf16;
typedef bf16  v16bf __attribute__((ext_vector_type(16)));
typedef bf16  v4bf  __attribute__((ext_vector_type(4)));
typedef float v8f   __attribute__((ext_vector_type(8)));

__device__ __forceinline__ const float* row_ptr(const float* __restrict__ x,
                                                const float* __restrict__ fwd,
                                                const float* __restrict__ bwd,
                                                int m) {
    int b = m / TT;
    int t = m - b * TT;
    if (t == 0)      return fwd + (size_t)b * DD;
    if (t == TT - 1) return bwd + (size_t)b * DD;
    return x + ((size_t)b * SS + (size_t)(t - 1)) * DD;
}

// ---------------------------------------------------------------------------
// Pre-pass: convert [wq|wk|wv] into WMMA-B-fragment-ordered bf16 hi/lo blocks.
// Block layout per (wt, kstep): lane l owns 64B: [hi v16bf][lo v16bf].
// Out-of-range columns (n >= 100) are written as zero -> no guards in GEMM.
// ---------------------------------------------------------------------------
__global__ __launch_bounds__(256) void prep_b_kernel(
    const float* __restrict__ wq, const float* __restrict__ wk,
    const float* __restrict__ wv, unsigned int* __restrict__ bws)
{
    const int lane = threadIdx.x & 31;
    const int frag = blockIdx.x * 8 + (threadIdx.x >> 5);  // 0 .. 21*256-1
    const int wt   = frag >> 8;                            // 0..20
    const int kb   = frag & 255;                           // K-step
    const int wsel  = wt / NT;
    const int ntile = wt - wsel * NT;
    const float* wmat = (wsel == 0) ? wq : (wsel == 1) ? wk : wv;

    const int  n   = ntile * 16 + (lane & 15);
    const bool nok = n < PP;
    const int  khalf = (lane >> 4) * 16;   // B 32x16: lanes 16-31 hold K+16

    v16bf hi, lo;
    #pragma unroll
    for (int r = 0; r < 8; ++r) {
        int k = kb * 32 + khalf + 2 * r;
        float b0 = nok ? wmat[(size_t)k * PP + n]       : 0.0f;
        float b1 = nok ? wmat[(size_t)(k + 1) * PP + n] : 0.0f;
        bf16 h0 = (bf16)b0, h1 = (bf16)b1;
        hi[2 * r]     = h0;
        hi[2 * r + 1] = h1;
        lo[2 * r]     = (bf16)(b0 - (float)h0);
        lo[2 * r + 1] = (bf16)(b1 - (float)h1);
    }
    unsigned int* dst = bws + (size_t)frag * BFRAG_DW + lane * 16;
    *(v16bf*)dst       = hi;
    *(v16bf*)(dst + 8) = lo;
}

// Convert one fp32 float4 to bf16 hi/lo pairs and store into LDS fragments.
__device__ __forceinline__ void stage_slice(unsigned int* d, float4 a) {
    bf16 h0 = (bf16)a.x, h1 = (bf16)a.y, h2 = (bf16)a.z, h3 = (bf16)a.w;
    v4bf hp; hp[0] = h0; hp[1] = h1; hp[2] = h2; hp[3] = h3;
    v4bf lp;
    lp[0] = (bf16)(a.x - (float)h0);
    lp[1] = (bf16)(a.y - (float)h1);
    lp[2] = (bf16)(a.z - (float)h2);
    lp[3] = (bf16)(a.w - (float)h3);
    *(v4bf*)d       = hp;
    *(v4bf*)(d + 8) = lp;
}

// ---------------------------------------------------------------------------
// GEMM: grid (51 M-pairs, 3 weights), 256 threads = 8 waves.
// All 8 waves cooperatively stage a 32row x 32K fp32 slice as bf16 hi/lo
// A-fragments in LDS (double buffered, split barrier / K-step). Waves 0..6
// each own one N-tile and run the 3-term bf16-split WMMA chain on 2 M-tiles.
// ---------------------------------------------------------------------------
__global__ __launch_bounds__(256) void proj_gemm_kernel(
    const float* __restrict__ x,  const float* __restrict__ fwd,
    const float* __restrict__ bwd, const unsigned int* __restrict__ bws,
    float* __restrict__ proj)
{
    __shared__ __attribute__((aligned(32))) unsigned int abuf[2048]; // 2 buf x 2 tiles x 512 dw

    const int tid  = threadIdx.x;
    const int lane = tid & 31;
    const int wave = tid >> 5;        // 0..7 (wave 7: staging only)
    const int wsel = blockIdx.y;      // weight select
    const int mt0  = blockIdx.x * 2;  // two consecutive M-tiles

    // wave id as a proven-uniform SGPR value -> scalar branch, no exec masking
    const int  wave_u = __builtin_amdgcn_readfirstlane(tid) >> 5;
    const bool comp   = wave_u < NT;

    // -- producer mapping: thread -> one float4 of the 32x32 slice --
    const int prow = tid >> 3;                 // 0..31 (row in slice)
    const int kq   = (tid & 7) * 4;            // 0,4,...,28
    const float4* gsrc =
        (const float4*)row_ptr(x, fwd, bwd, mt0 * 16 + prow) + (kq >> 2);
    const int khsel    = kq >> 4;
    const int lane_dst = khsel * 16 + (prow & 15);
    const int r0       = (kq & 15) >> 1;  // even
    const int pdst     = (prow >> 4) * 512 + lane_dst * 16 + r0;

    // -- consumer setup --
    const unsigned int* bptr =
        bws + (size_t)((wsel * NT + (comp ? wave : 0)) * KSTEPS) * BFRAG_DW + lane * 16;

    v8f acc0 = {}, acc1 = {};

    float4 areg = gsrc[0];   // K-slice 0
    #pragma unroll 2         // buffer offsets become immediates
    for (int i = 0; i < KSTEPS; ++i) {
        stage_slice(&abuf[(i & 1) * 1024 + pdst], areg);

        // split barrier: signal, overlap next global A load, then wait
        __asm__ volatile("s_wait_dscnt 0x0\n\ts_barrier_signal -1" ::: "memory");
        if (i + 1 < KSTEPS) areg = gsrc[(i + 1) * 8];
        __asm__ volatile("s_barrier_wait -1" ::: "memory");

        if (comp) {
            const unsigned int* bp = bptr + (size_t)i * BFRAG_DW;
            __builtin_prefetch(bp + 4 * BFRAG_DW, 0, 3);
            v16bf bh = *(const v16bf*)bp;
            v16bf bl = *(const v16bf*)(bp + 8);
            const unsigned int* a0 = &abuf[(i & 1) * 1024 + lane * 16];
            v16bf a0h = *(const v16bf*)a0;
            v16bf a0l = *(const v16bf*)(a0 + 8);
            v16bf a1h = *(const v16bf*)(a0 + 512);
            v16bf a1l = *(const v16bf*)(a0 + 520);
            acc0 = __builtin_amdgcn_wmma_f32_16x16x32_bf16(false, a0h, false, bh,
                                                           (short)0, acc0, false, false);
            acc0 = __builtin_amdgcn_wmma_f32_16x16x32_bf16(false, a0h, false, bl,
                                                           (short)0, acc0, false, false);
            acc0 = __builtin_amdgcn_wmma_f32_16x16x32_bf16(false, a0l, false, bh,
                                                           (short)0, acc0, false, false);
            acc1 = __builtin_amdgcn_wmma_f32_16x16x32_bf16(false, a1h, false, bh,
                                                           (short)0, acc1, false, false);
            acc1 = __builtin_amdgcn_wmma_f32_16x16x32_bf16(false, a1h, false, bl,
                                                           (short)0, acc1, false, false);
            acc1 = __builtin_amdgcn_wmma_f32_16x16x32_bf16(false, a1l, false, bh,
                                                           (short)0, acc1, false, false);
        }
    }

    if (comp) {
        int ncol = wave * 16 + (lane & 15);
        if (ncol < PP) {
            float* outp = proj + (size_t)wsel * ROWS * PP;
            int mb0 = mt0 * 16 + 8 * (lane >> 4);   // C/D: reg j -> M=j+8*(lane>=16)
            int mb1 = mb0 + 16;
            #pragma unroll
            for (int j = 0; j < 8; ++j) {
                outp[(size_t)(mb0 + j) * PP + ncol] = acc0[j];
                outp[(size_t)(mb1 + j) * PP + ncol] = acc1[j];
            }
        }
    }
}

// One wave32 per (b,s): 3 dots of length 100, softmax over 3, weighted V sum.
__global__ __launch_bounds__(128) void local_attn_kernel(
    const float* __restrict__ proj, float* __restrict__ out)
{
    const int lane = threadIdx.x & 31;
    const int wid  = threadIdx.x >> 5;
    const int idx  = blockIdx.x * 4 + wid;
    if (idx >= BB * SS) return;

    const int b = idx / SS;
    const int s = idx - b * SS;

    const float* Q = proj;
    const float* K = proj + (size_t)ROWS * PP;
    const float* V = proj + (size_t)2 * ROWS * PP;
    const float* qrow = Q + (size_t)(b * TT + s + 1) * PP;
    const float* krow = K + (size_t)(b * TT + s) * PP;
    const float* vrow = V + (size_t)(b * TT + s) * PP;

    float d0 = 0.f, d1 = 0.f, d2 = 0.f;
    for (int p = lane; p < PP; p += 32) {
        float qv = qrow[p];
        d0 += qv * krow[p];
        d1 += qv * krow[PP + p];
        d2 += qv * krow[2 * PP + p];
    }
    #pragma unroll
    for (int off = 16; off > 0; off >>= 1) {
        d0 += __shfl_xor(d0, off, 32);
        d1 += __shfl_xor(d1, off, 32);
        d2 += __shfl_xor(d2, off, 32);
    }
    float mx = fmaxf(d0, fmaxf(d1, d2));
    float e0 = __expf(d0 - mx), e1 = __expf(d1 - mx), e2 = __expf(d2 - mx);
    float inv = 1.0f / (e0 + e1 + e2);
    e0 *= inv; e1 *= inv; e2 *= inv;

    for (int p = lane; p < PP; p += 32)
        out[(size_t)idx * PP + p] = e0 * vrow[p] + e1 * vrow[PP + p] + e2 * vrow[2 * PP + p];
}

extern "C" void kernel_launch(void* const* d_in, const int* in_sizes, int n_in,
                              void* d_out, int out_size, void* d_ws, size_t ws_size,
                              hipStream_t stream)
{
    (void)in_sizes; (void)n_in; (void)out_size; (void)ws_size;
    const float* x   = (const float*)d_in[0];
    const float* wq  = (const float*)d_in[1];
    const float* wk  = (const float*)d_in[2];
    const float* wv  = (const float*)d_in[3];
    const float* fwd = (const float*)d_in[4];
    const float* bwd = (const float*)d_in[5];

    unsigned int* bws = (unsigned int*)d_ws;           // 10.5 MB B fragments
    float* proj = (float*)d_ws + BWS_DW;               // 3 * 1632 * 100 fp32

    prep_b_kernel<<<dim3(NWT * KSTEPS / 8), dim3(256), 0, stream>>>(wq, wk, wv, bws);
    proj_gemm_kernel<<<dim3(ROWS / 32, 3), dim3(256), 0, stream>>>(
        x, fwd, bwd, bws, proj);
    local_attn_kernel<<<dim3((BB * SS + 3) / 4), dim3(128), 0, stream>>>(
        proj, (float*)d_out);
}
